// TransitionStepMixin_88699664597041
// MI455X (gfx1250) — matrix operand order repaired and checked by reading, not compiled, not run on hardware
//
#include <hip/hip_runtime.h>
#include <hip/hip_bf16.h>
#include <stdint.h>

typedef __attribute__((ext_vector_type(16))) _Float16 v16h;
typedef __attribute__((ext_vector_type(2)))  _Float16 h2;
typedef __attribute__((ext_vector_type(8)))  float    v8f;
typedef __attribute__((ext_vector_type(8)))  int      v8i;

#define B_  256
#define H_  2048
#define S_  64
#define TILE_B 16
#define PITCH  68   // padded dwords per LDS row (avoids bank conflicts)

// Cross-half (lane ^ 16) exchange as pure VALU: v_permlanex16_b32 with
// identity lane selects. No LDS round-trip, no dscnt wait on the serial chain.
__device__ inline int permx16_i(int v) {
  return __builtin_amdgcn_permlanex16(v, v, 0x76543210, 0xfedcba98, false, false);
}
__device__ inline float permx16_f(float v) {
  int b = __builtin_bit_cast(int, v);
  return __builtin_bit_cast(float, permx16_i(b));
}

// One wave (32 lanes) owns a 16-batch tile.
//   State pred^T kept in 4 WMMA C-fragments acc[mt] (v8f):
//     lane l, vgpr r, tile mt  <->  state s = 16*mt + 8*(l/16) + r, batch = b0 + (l%16)
//   pred_next^T = T^T @ est^T via 8x v_wmma_f32_16x16x32_f16 per step
//     (A = T^T fragments, constant in regs; B = est^T built in-lane from
//      packed half2 words, no lane exchange)
__global__ __launch_bounds__(32) void hmm_filter_kernel(
    const float* __restrict__ lik,    // [B,H,S]
    const float* __restrict__ init,   // [S]
    const float* __restrict__ Tm,     // [S,S] row-stochastic
    float* __restrict__ est_out,      // [B,H,S]
    float* __restrict__ pred_out)     // [B,H,S]
{
  __shared__ float lds[2][TILE_B][PITCH];

  const int lane = threadIdx.x;   // 0..31
  const int col  = lane & 15;     // batch-in-tile (N index)
  const int hl   = lane >> 4;     // which 16-lane half
  const int b0   = blockIdx.x * TILE_B;

  // ---------- A = T^T fragments (constant across all 2048 steps) ----------
  // A[mt][m][k] = T[k_global][16*mt + m], m = col.
  // 16-bit A 16x32 layout: lane l elem e -> K_local = 16*(e/8) + 8*(l/16) + (e%8)
  v16h Afrag[2][4];  // [kc][mt]
#pragma unroll
  for (int kc = 0; kc < 2; ++kc)
#pragma unroll
    for (int mt = 0; mt < 4; ++mt)
#pragma unroll
      for (int e = 0; e < 16; ++e) {
        int K = 32 * kc + 16 * (e >> 3) + 8 * hl + (e & 7);
        Afrag[kc][mt][e] = (_Float16)Tm[K * S_ + 16 * mt + col];
      }

  // ---------- pred0 = init @ T (same for every batch -> broadcast) ----------
  v8f acc[4];
#pragma unroll
  for (int mt = 0; mt < 4; ++mt)
#pragma unroll
    for (int r = 0; r < 8; ++r) acc[mt][r] = 0.0f;
  for (int k = 0; k < S_; ++k) {
    float ik = init[k];
#pragma unroll
    for (int mt = 0; mt < 4; ++mt)
#pragma unroll
      for (int r = 0; r < 8; ++r)
        acc[mt][r] = fmaf(ik, Tm[k * S_ + 16 * mt + 8 * hl + r], acc[mt][r]);
  }

  // ---------- async-to-LDS stage of one lik tile [16 x 64] (4 KB) ----------
  auto stage = [&](int buf, int h) {
#pragma unroll
    for (int i = 0; i < 8; ++i) {
      int rr = 2 * i + hl;  // row (batch-in-tile) covered by this lane
      const float* gp =
          lik + ((size_t)(b0 + rr) * H_ + (size_t)h) * S_ + 4 * col;
      unsigned loff = (unsigned)(uintptr_t)&lds[buf][rr][4 * col];
      asm volatile("global_load_async_to_lds_b128 %0, %1, off"
                   :: "v"(loff), "v"((unsigned long long)(uintptr_t)gp)
                   : "memory");
    }
  };

  stage(0, 0);  // prologue prefetch

  const size_t strideBH = (size_t)H_ * S_;

  for (int h = 0; h < H_; ++h) {
    const int cur = h & 1;

    // wait for this step's lik tile
    asm volatile("s_wait_asynccnt 0x0" ::: "memory");

    // read lik^T for our 32 (mt,r) states: 8 consecutive floats per mt
    float likv[4][8];
#pragma unroll
    for (int mt = 0; mt < 4; ++mt) {
      const float4* p =
          reinterpret_cast<const float4*>(&lds[cur][col][16 * mt + 8 * hl]);
      float4 a = p[0], b = p[1];
      likv[mt][0] = a.x; likv[mt][1] = a.y; likv[mt][2] = a.z; likv[mt][3] = a.w;
      likv[mt][4] = b.x; likv[mt][5] = b.y; likv[mt][6] = b.z; likv[mt][7] = b.w;
    }

    // prefetch next step's tile into the other buffer (overlaps with compute)
    int hn = (h + 1 < H_) ? h + 1 : H_ - 1;
    stage(cur ^ 1, hn);

    // ----- update: est = L1normalize(pred * lik) -----
    float est[4][8];
    float ssum = 0.0f;
#pragma unroll
    for (int mt = 0; mt < 4; ++mt)
#pragma unroll
      for (int r = 0; r < 8; ++r) {
        float v = acc[mt][r] * likv[mt][r];
        est[mt][r] = v;
        ssum += __builtin_fabsf(v);
      }
    // lane + partner(lane^16) together hold all 64 states of this batch
    ssum += permx16_f(ssum);
    float inv = 1.0f / fmaxf(ssum, 1e-12f);
#pragma unroll
    for (int mt = 0; mt < 4; ++mt)
#pragma unroll
      for (int r = 0; r < 8; ++r) est[mt][r] *= inv;

    // store est_traj[b][h][:]
#pragma unroll
    for (int mt = 0; mt < 4; ++mt) {
      size_t off = (size_t)(b0 + col) * strideBH + (size_t)h * S_ +
                   16 * mt + 8 * hl;
      float4 lo = make_float4(est[mt][0], est[mt][1], est[mt][2], est[mt][3]);
      float4 hi = make_float4(est[mt][4], est[mt][5], est[mt][6], est[mt][7]);
      *reinterpret_cast<float4*>(est_out + off)     = lo;
      *reinterpret_cast<float4*>(est_out + off + 4) = hi;
    }

    // ----- pack est rows to half2 words (fuses to v_cvt_pk_f16_f32) -----
    int pk[4][4];
#pragma unroll
    for (int mt = 0; mt < 4; ++mt)
#pragma unroll
      for (int j = 0; j < 4; ++j) {
        h2 p;
        p[0] = (_Float16)est[mt][2 * j];
        p[1] = (_Float16)est[mt][2 * j + 1];
        pk[mt][j] = __builtin_bit_cast(int, p);
      }

    // ----- build B = est^T fragments (32x16 f16 each) -----
    // B[kc] lane l elem e -> state = 32*kc + 16*(e/8) + 8*(l/16) + (e%8), batch = l%16.
    // Every lane already holds its own elements; no cross-half exchange needed.
    v16h Bfrag[2];
#pragma unroll
    for (int kc = 0; kc < 2; ++kc) {
      v8i bb;
#pragma unroll
      for (int j = 0; j < 4; ++j) {
        bb[j]     = pk[2 * kc][j];
        bb[4 + j] = pk[2 * kc + 1][j];
      }
      Bfrag[kc] = __builtin_bit_cast(v16h, bb);
    }

    // ----- pred_next^T = T^T @ est^T : 8 WMMAs, f32 accumulate -----
#pragma unroll
    for (int mt = 0; mt < 4; ++mt) {
      v8f c = {0.f, 0.f, 0.f, 0.f, 0.f, 0.f, 0.f, 0.f};
      c = __builtin_amdgcn_wmma_f32_16x16x32_f16(
              false, Afrag[0][mt], false, Bfrag[0], (short)0, c, false, false);
      c = __builtin_amdgcn_wmma_f32_16x16x32_f16(
              false, Afrag[1][mt], false, Bfrag[1], (short)0, c, false, false);
      acc[mt] = c;
    }

    // store pred_traj[b][h][:] (prediction produced at this step)
#pragma unroll
    for (int mt = 0; mt < 4; ++mt) {
      size_t off = (size_t)(b0 + col) * strideBH + (size_t)h * S_ +
                   16 * mt + 8 * hl;
      float4 lo = make_float4(acc[mt][0], acc[mt][1], acc[mt][2], acc[mt][3]);
      float4 hi = make_float4(acc[mt][4], acc[mt][5], acc[mt][6], acc[mt][7]);
      *reinterpret_cast<float4*>(pred_out + off)     = lo;
      *reinterpret_cast<float4*>(pred_out + off + 4) = hi;
    }
  }
}

extern "C" void kernel_launch(void* const* d_in, const int* in_sizes, int n_in,
                              void* d_out, int out_size, void* d_ws, size_t ws_size,
                              hipStream_t stream) {
  const float* lik  = (const float*)d_in[0];  // [B,H,S]
  const float* init = (const float*)d_in[1];  // [S]
  const float* Tm   = (const float*)d_in[2];  // [S,S]
  float* est_out  = (float*)d_out;                           // [B,H,S]
  float* pred_out = est_out + (size_t)B_ * H_ * S_;          // [B,H,S]

  dim3 grid(B_ / TILE_B);  // 16 waves, one 16-batch tile each
  dim3 block(32);
  hmm_filter_kernel<<<grid, block, 0, stream>>>(lik, init, Tm, est_out, pred_out);
}
